// MultiHeadAttention_4896262717595
// MI455X (gfx1250) — matrix operand order repaired and checked by reading, not compiled, not run on hardware
//
#include <hip/hip_runtime.h>
#include <math.h>

// ---------------------------------------------------------------------------
// Cosine-similarity multi-head attention for MI455X (gfx1250, wave32, WMMA).
//
// Roofline: 44 GFLOP, mandatory HBM traffic ~590MB (att output alone is
// 328MB) -> ~25us floor at 23.3TB/s. F32 16x16x4 WMMA would make this
// compute-bound; V_WMMA_F32_16X16X32_F16 (8x K per instruction, fp32
// accumulate) keeps it memory-bound. Q/K/V/ctx intermediates stored f16 to
// halve workspace round-trip traffic (1.05GB -> 525MB). All reductions,
// norms, softmax and both outputs (out, att) remain fp32. Score/softmax
// reciprocals use v_rcp_f32 (not IEEE div) to keep the epilogue VALU-light.
//
// Pipeline:
//   0) convert the 4 weight matrices fp32 -> f16 (256KB total, L2-resident)
//   1-3) Q/K/V projections: f16 WMMA GEMM, f16 outputs
//   4) fused per-(b,h) attention: q@k^T (f16 WMMA, 1 inst/tile), cosine
//      normalization + pos_bias (fp32, rcp+fma), softmax (wave32 shuffle
//      reduce, fp32), att@v (f16 WMMA, K padded 200->224 with zeros)
//   5) out = ctx @ R_w^T + R_b : f16 WMMA, fp32 output
// ---------------------------------------------------------------------------

#define EMB   128
#define NT    50
#define NC    200
#define NUMH  8
#define BATCH 1024

typedef __attribute__((ext_vector_type(8)))  float     v8f;
typedef _Float16 __attribute__((ext_vector_type(8)))   h8;
typedef _Float16 __attribute__((ext_vector_type(16)))  v16h;

// D(16x16,f32) += A(16xK,f16) * B(Kx16,f16), B staged transposed.
//  As: 16 rows (M) x K halves, row-major, stride sA (halves, multiple of 8)
//  Bs: 16 rows (N) x K halves, row-major, stride sB (i.e. B transposed)
// f16 16x16x32 operand layout (ISA 7.12.2):
//  A: lane r=lane&15 holds row r; halves = A[r][c1..c1+7] ++ A[r][c1+16..+23]
//     with c1 = kk + 8*(lane>>4)            (two 16B chunks)
//  B: lane c=lane&15 holds col c; halves = Bt[c][c1..c1+7] ++ Bt[c][c1+16..+23]
//     with c1 = kk + 8*(lane>>4)            (two 16B chunks, same k order as A)
//  C/D: VGPR r holds D[r + 8*(lane>>4)][lane&15]
__device__ __forceinline__ v8f wmma_tile_f16(const _Float16* __restrict__ As, int sA,
                                             const _Float16* __restrict__ Bs, int sB,
                                             int K, int lane) {
  v8f acc = {0.f, 0.f, 0.f, 0.f, 0.f, 0.f, 0.f, 0.f};
  const int r  = lane & 15;
  const int hi = lane >> 4;                  // 0 or 1
  for (int kk = 0; kk < K; kk += 32) {
    const _Float16* ap = As + r * sA + kk + hi * 8;
    const h8 alo = *(const h8*)(ap);
    const h8 ahi = *(const h8*)(ap + 16);
    const v16h a = __builtin_shufflevector(alo, ahi,
        0, 1, 2, 3, 4, 5, 6, 7, 8, 9, 10, 11, 12, 13, 14, 15);
    const _Float16* bp = Bs + r * sB + kk + hi * 8;
    const h8 blo = *(const h8*)(bp);
    const h8 bhi = *(const h8*)(bp + 16);
    const v16h b = __builtin_shufflevector(blo, bhi,
        0, 1, 2, 3, 4, 5, 6, 7, 8, 9, 10, 11, 12, 13, 14, 15);
    acc = __builtin_amdgcn_wmma_f32_16x16x32_f16(false, a, false, b,
                                                 (short)0, acc, false, false);
  }
  return acc;
}

__global__ __launch_bounds__(256) void cvt_f32_to_f16_kernel(
    const float* __restrict__ src, _Float16* __restrict__ dst, int n) {
  const int i = blockIdx.x * 256 + threadIdx.x;
  if (i < n) dst[i] = (_Float16)src[i];
}

// Y[M][N] = X[M][K] @ W16[N][K]^T + bias[N]
// grid = (M/16, N/128), block = 256 (8 waves, one 16x16 tile each).
// IN_F16: X given as f16 (Xh) instead of fp32 (Xf).
// OUT_F32: store fp32 (Yf) instead of f16 (Yh).
template <bool IN_F16, bool OUT_F32>
__global__ __launch_bounds__(256) void gemm_bias_f16_kernel(
    const float* __restrict__ Xf, const _Float16* __restrict__ Xh,
    const _Float16* __restrict__ W16, const float* __restrict__ bias,
    float* __restrict__ Yf, _Float16* __restrict__ Yh,
    int M, int N, int K) {
  __shared__ __align__(16) _Float16 Xs[16 * 264];  // K<=256, stride K+8 halves
  const int tid  = threadIdx.x;
  const int wave = tid >> 5;
  const int lane = tid & 31;
  const int sA   = K + 8;                          // 16B-aligned rows
  const int row0 = blockIdx.x * 16;

  for (int i = tid; i < 16 * K; i += 256) {
    const int r = i / K, c = i - r * K;
    const size_t gidx = (size_t)(row0 + r) * K + c;
    Xs[r * sA + c] = IN_F16 ? Xh[gidx] : (_Float16)Xf[gidx];
  }
  __syncthreads();

  const int n0 = blockIdx.y * 128 + wave * 16;
  // W16 is (N,K) row-major f16 == transposed-B layout; read via b128 loads.
  v8f acc = wmma_tile_f16(Xs, sA, W16 + (size_t)n0 * K, K, K, lane);

  const int n  = n0 + (lane & 15);
  const int mo = (lane >> 4) << 3;                 // 0 or 8
  const float bn = bias[n];
#pragma unroll
  for (int r = 0; r < 8; ++r) {
    const size_t idx = (size_t)(row0 + r + mo) * N + n;
    const float y = acc[r] + bn;
    if (OUT_F32) Yf[idx] = y; else Yh[idx] = (_Float16)y;
  }
}

// Fused per-(b,h) attention. grid = BATCH*NUMH, 256 threads,
// dynamic LDS = 121,984 B (fits CDNA5's 320KB/WGP; 2 workgroups per WGP).
__global__ __launch_bounds__(256) void attn_kernel(
    const _Float16* __restrict__ Q16, const _Float16* __restrict__ K16,
    const _Float16* __restrict__ V16, const float* __restrict__ pos_bias,
    _Float16* __restrict__ ctx16, float* __restrict__ att_out) {
  extern __shared__ float smem[];
  float* pb_s  = smem;                         // 208   pos_bias (padded)
  float* qn_s  = pb_s + 208;                   // 64    |q| per padded row
  float* kn_s  = qn_s + 64;                    // 208   |k| per padded col
  float* att_s = kn_s + 208;                   // 64 x 210 fp32 scores/exp
  _Float16* qs = (_Float16*)(att_s + 64 * 210);  // 64  x 32, stride 40
  _Float16* ks = qs + 64 * 40;                   // 208 x 32, stride 40
  _Float16* vT = ks + 208 * 40;                  // 32  x 224, stride 232
  _Float16* a16 = vT + 32 * 232;                 // 64  x 224, stride 232

  const int tid  = threadIdx.x;
  const int wave = tid >> 5;
  const int lane = tid & 31;
  const int bh = blockIdx.x;
  const int bb = bh >> 3;
  const int h  = bh & 7;
  const size_t qbase = (size_t)bb * NT * 256 + h * 32;
  const size_t cbase = (size_t)bb * NC * 256 + h * 32;

  for (int i = tid; i < 208; i += 256) pb_s[i] = (i < NC) ? pos_bias[i] : 0.f;
  for (int i = tid; i < 64 * 32; i += 256) {
    const int t = i >> 5, d = i & 31;
    qs[t * 40 + d] = (t < NT) ? Q16[qbase + (size_t)t * 256 + d] : (_Float16)0.f;
  }
  for (int i = tid; i < 224 * 32; i += 256) {
    const int c = i >> 5, d = i & 31;
    const bool ok = (c < NC);
    const _Float16 kh = ok ? K16[cbase + (size_t)c * 256 + d] : (_Float16)0.f;
    const _Float16 vh = ok ? V16[cbase + (size_t)c * 256 + d] : (_Float16)0.f;
    if (c < 208) ks[c * 40 + d] = kh;
    vT[d * 232 + c] = vh;                      // zero-padded K cols 200..223
  }
  __syncthreads();

  // fp32 row norms of the (f16-rounded) q/k vectors
  for (int i = tid; i < 64 + 208; i += 256) {
    if (i < 64) {
      float s = 0.f;
#pragma unroll
      for (int d = 0; d < 32; ++d) { const float x = (float)qs[i * 40 + d]; s += x * x; }
      qn_s[i] = sqrtf(s);
    } else {
      const int c = i - 64;
      float s = 0.f;
#pragma unroll
      for (int d = 0; d < 32; ++d) { const float x = (float)ks[c * 40 + d]; s += x * x; }
      kn_s[c] = sqrtf(s);
    }
  }
  __syncthreads();

  // scores: 4 row-tiles x 13 col-tiles, one v_wmma_f32_16x16x32_f16 each.
  // Epilogue: v_rcp_f32 + fma per element (no IEEE div).
  for (int tile = wave; tile < 52; tile += 8) {
    const int tm = tile & 3, tn = tile >> 2;
    v8f acc = wmma_tile_f16(qs + tm * 16 * 40, 40, ks + tn * 16 * 40, 40, 32, lane);
    const int n  = tn * 16 + (lane & 15);
    const int mo = (lane >> 4) << 3;
    if (n < NC) {
      const float kn = kn_s[n];
      const float pb = pb_s[n];
#pragma unroll
      for (int r = 0; r < 8; ++r) {
        const int m = tm * 16 + r + mo;
        const float rden = __builtin_amdgcn_rcpf(fmaxf(qn_s[m] * kn, 1e-6f));
        att_s[m * 210 + n] = fmaf(acc[r], rden, pb);
      }
    }
  }
  __syncthreads();

  // softmax (fp32) per row; probabilities -> fp32 global att + f16 LDS a16
  for (int t = wave; t < 64; t += 8) {
    if (t < NT) {
      float* row = att_s + t * 210;
      float mx = -3.0e38f;
      for (int c = lane; c < NC; c += 32) mx = fmaxf(mx, row[c]);
#pragma unroll
      for (int o = 16; o > 0; o >>= 1) mx = fmaxf(mx, __shfl_xor(mx, o, 32));
      float sum = 0.f;
      for (int c = lane; c < NC; c += 32) {
        const float e = __expf(row[c] - mx);
        row[c] = e;
        sum += e;
      }
#pragma unroll
      for (int o = 16; o > 0; o >>= 1) sum += __shfl_xor(sum, o, 32);
      const float inv = __builtin_amdgcn_rcpf(sum);
      for (int c = lane; c < 224; c += 32) {
        const float p = (c < NC) ? row[c] * inv : 0.f;
        a16[t * 232 + c] = (_Float16)p;
        if (c < NC) att_out[((size_t)bh * NT + t) * NC + c] = p;
      }
    } else {
      for (int c = lane; c < 224; c += 32) a16[t * 232 + c] = (_Float16)0.f;
    }
  }
  __syncthreads();

  // ctx = att @ v : 4 row-tiles x 2 col-tiles, one tile per wave, K=224
  {
    const int tm = wave & 3, tn = wave >> 2;
    v8f acc = wmma_tile_f16(a16 + tm * 16 * 232, 232, vT + tn * 16 * 232, 232, 224, lane);
    const int d  = tn * 16 + (lane & 15);
    const int mo = (lane >> 4) << 3;
#pragma unroll
    for (int r = 0; r < 8; ++r) {
      const int t = tm * 16 + r + mo;
      if (t < NT) ctx16[((size_t)bb * NT + t) * 256 + h * 32 + d] = (_Float16)acc[r];
    }
  }
}

extern "C" void kernel_launch(void* const* d_in, const int* in_sizes, int n_in,
                              void* d_out, int out_size, void* d_ws, size_t ws_size,
                              hipStream_t stream) {
  (void)in_sizes; (void)n_in; (void)out_size; (void)ws_size;
  const float* queries  = (const float*)d_in[0];
  const float* keys     = (const float*)d_in[1];
  const float* values   = (const float*)d_in[2];
  const float* At_w     = (const float*)d_in[3];
  const float* At_b     = (const float*)d_in[4];
  const float* Ac_w     = (const float*)d_in[5];
  const float* Ac_b     = (const float*)d_in[6];
  const float* Bc_w     = (const float*)d_in[7];
  const float* Bc_b     = (const float*)d_in[8];
  const float* pos_bias = (const float*)d_in[9];
  const float* R_w      = (const float*)d_in[10];
  const float* R_b      = (const float*)d_in[11];

  const int Mq = BATCH * NT;   // 51200
  const int Mc = BATCH * NC;   // 204800
  const int WN = 256 * 128;    // elements per projection weight (= R weight)

  // Workspace (all f16, every segment 16B-aligned):
  //   Wq|Wk|Wv|Wr (4 x 64KB) | Q16 26.2MB | K16 104.9MB | V16 104.9MB | ctx16 26.2MB
  _Float16* Wq16 = (_Float16*)d_ws;
  _Float16* Wk16 = Wq16 + WN;
  _Float16* Wv16 = Wk16 + WN;
  _Float16* Wr16 = Wv16 + WN;
  _Float16* Q16  = Wr16 + WN;
  _Float16* K16  = Q16 + (size_t)Mq * 256;
  _Float16* V16  = K16 + (size_t)Mc * 256;
  _Float16* C16  = V16 + (size_t)Mc * 256;

  float* out = (float*)d_out;               // tuple elem 0: (1024,50,128)
  float* att = out + (size_t)Mq * EMB;      // tuple elem 1: (1024,8,50,200)

  const int cvb = (WN + 255) / 256;
  cvt_f32_to_f16_kernel<<<cvb, 256, 0, stream>>>(At_w, Wq16, WN);
  cvt_f32_to_f16_kernel<<<cvb, 256, 0, stream>>>(Ac_w, Wk16, WN);
  cvt_f32_to_f16_kernel<<<cvb, 256, 0, stream>>>(Bc_w, Wv16, WN);
  cvt_f32_to_f16_kernel<<<cvb, 256, 0, stream>>>(R_w,  Wr16, WN);

  gemm_bias_f16_kernel<false, false><<<dim3(Mq / 16, 2), 256, 0, stream>>>(
      queries, nullptr, Wq16, At_b, nullptr, Q16, Mq, 256, 128);
  gemm_bias_f16_kernel<false, false><<<dim3(Mc / 16, 2), 256, 0, stream>>>(
      keys, nullptr, Wk16, Ac_b, nullptr, K16, Mc, 256, 128);
  gemm_bias_f16_kernel<false, false><<<dim3(Mc / 16, 2), 256, 0, stream>>>(
      values, nullptr, Wv16, Bc_b, nullptr, V16, Mc, 256, 128);

  const size_t smem_bytes =
      (size_t)(208 + 64 + 208 + 64 * 210) * sizeof(float) +
      (size_t)(64 * 40 + 208 * 40 + 32 * 232 + 64 * 232) * sizeof(_Float16);
  attn_kernel<<<dim3(BATCH * NUMH), 256, smem_bytes, stream>>>(
      Q16, K16, V16, pos_bias, C16, att);

  gemm_bias_f16_kernel<true, true><<<dim3(Mq / 16, 1), 256, 0, stream>>>(
      nullptr, C16, Wr16, R_b, out, nullptr, Mq, 128, 256);
}
